// SememeSumLstm_GRU_81999515615469
// MI455X (gfx1250) — compile-verified, matrix operand order checked by reading, not run on hardware
//
#include <hip/hip_runtime.h>
#include <cstdint>

// ---------------- CDNA5 WMMA types -------------------------------------------
typedef __attribute__((ext_vector_type(16))) __bf16 v16bf;
typedef __attribute__((ext_vector_type(8)))  __bf16 v8bf;
typedef __attribute__((ext_vector_type(8)))  float  v8f;

#define WMMA_BF16(a, b, c) \
  __builtin_amdgcn_wmma_f32_16x16x32_bf16(false, (a), false, (b), (short)0, (c), false, false)

// Problem constants (from reference)
#define TSTEPS 512
#define BATCH  128
#define INDIM  300
#define MEM    512
#define NROWS  (TSTEPS * BATCH)   // 65536
#define LDA_P  40                 // LDS pitch in bf16 (80B rows: 16B aligned, conflict-free)

static __device__ __forceinline__ float sigmoid_f(float x) { return 1.0f / (1.0f + __expf(-x)); }
static __device__ __forceinline__ float tanh_f(float x)    { return 2.0f / (1.0f + __expf(-2.0f * x)) - 1.0f; }

// A/B fragment for v_wmma_f32_16x16x32_bf16, built from fp32 memory (on-the-fly cvt).
// Lane layout (wave32): M(or N) = lane%16; element e<8 -> K = kb + 8*half + e,
// e>=8 -> K = kb + 16 + 8*half + (e-8).  Two contiguous 8-wide runs per lane.
static __device__ __forceinline__ v16bf frag_f32(const float* __restrict__ row, int kb, int lhalf) {
  const float* p0 = row + kb + 8 * lhalf;
  const float* p1 = row + kb + 16 + 8 * lhalf;
  v16bf v;
#pragma unroll
  for (int i = 0; i < 8; ++i) { v[i] = (__bf16)p0[i]; v[8 + i] = (__bf16)p1[i]; }
  return v;
}

static __device__ __forceinline__ v16bf frag_f32_guard(const float* __restrict__ row, int kb,
                                                       int lhalf, int kmax) {
  int k0 = kb + 8 * lhalf, k1 = kb + 16 + 8 * lhalf;
  v16bf v;
#pragma unroll
  for (int i = 0; i < 8; ++i) {
    v[i]     = (__bf16)((k0 + i < kmax) ? row[k0 + i] : 0.0f);
    v[8 + i] = (__bf16)((k1 + i < kmax) ? row[k1 + i] : 0.0f);
  }
  return v;
}

// Fragment from packed bf16 rows (global or LDS; 16B-aligned runs -> b128 loads).
static __device__ __forceinline__ v16bf frag_bf16(const __bf16* __restrict__ row, int kb, int lhalf) {
  v8bf lo = *(const v8bf*)(row + kb + 8 * lhalf);
  v8bf hi = *(const v8bf*)(row + kb + 16 + 8 * lhalf);
  v16bf v;
#pragma unroll
  for (int i = 0; i < 8; ++i) { v[i] = lo[i]; v[8 + i] = hi[i]; }
  return v;
}

// 16B async copy global -> LDS, tracked on ASYNCcnt (gfx1250 VGLOBAL encoding).
static __device__ __forceinline__ void async_copy_b128(unsigned lds_off, const void* gaddr) {
  asm volatile("global_load_async_to_lds_b128 %0, %1, off"
               :: "v"(lds_off), "v"(gaddr) : "memory");
}
static __device__ __forceinline__ void wait_async0() {
  asm volatile("s_wait_asynccnt 0x0" ::: "memory");
}

// ---------------- weight pack fp32 -> bf16 -----------------------------------
__global__ void __launch_bounds__(256) pack_bf16_k(const float* __restrict__ a,
                                                   __bf16* __restrict__ o, int n) {
  int i = blockIdx.x * 256 + threadIdx.x;
  if (i < n) o[i] = (__bf16)a[i];
}

// ---------------- Phase 1: input projections (bandwidth-bound GEMM) ----------
// Computes iou_x[65536][1024] (-> ws) and f_x[65536][512] (-> d_out overlay).
// Combined N = 1536 columns; K = 300 (9 full k-slabs of 32 + guarded tail).
__global__ void __launch_bounds__(256) gemm_input_k(
    const float* __restrict__ X,        // [65536][300]
    const float* __restrict__ W_ioux,   // [1024][300]
    const float* __restrict__ b_ioux,   // [1024]
    const float* __restrict__ W_fx,     // [512][300]
    const float* __restrict__ b_fx,     // [512]
    float* __restrict__ iou_x,          // [65536][1024]
    float* __restrict__ fx_out)         // [65536][512] == d_out overlay
{
  const int tid   = threadIdx.x;
  const int wave  = tid >> 5;
  const int lane  = tid & 31;
  const int lhalf = lane >> 4;
  const int lrow  = lane & 15;

  const long gw      = (long)blockIdx.x * 8 + wave;   // one 16x16 tile per wave
  const int  rowTile = (int)(gw / 96);                // 4096 row tiles
  const int  colTile = (int)(gw % 96);                // 96 col tiles (1536 cols)
  const int  mbase   = rowTile * 16;
  const int  nbase   = colTile * 16;
  const int  gn      = nbase + lrow;

  const float* xr = X + (size_t)(mbase + lrow) * INDIM;
  const float* wr = (gn < 1024) ? (W_ioux + (size_t)gn * INDIM)
                                : (W_fx + (size_t)(gn - 1024) * INDIM);

  v8f acc = {};
#pragma unroll 3
  for (int kb = 0; kb < 288; kb += 32) {              // fully in-bounds slabs
    v16bf a = frag_f32(xr, kb, lhalf);
    v16bf b = frag_f32(wr, kb, lhalf);
    acc = WMMA_BF16(a, b, acc);
  }
  {                                                   // tail slab K=288..319, guard at 300
    v16bf a = frag_f32_guard(xr, 288, lhalf, INDIM);
    v16bf b = frag_f32_guard(wr, 288, lhalf, INDIM);
    acc = WMMA_BF16(a, b, acc);
  }

  const float bias = (gn < 1024) ? b_ioux[gn] : b_fx[gn - 1024];
#pragma unroll
  for (int r = 0; r < 8; ++r) {
    const int   gm = mbase + r + 8 * lhalf;           // C layout: M = r + 8*half, N = lane%16
    const float v  = acc[r] + bias;
    if (gn < 1024) iou_x[(size_t)gm * 1024 + gn] = v;
    else           fx_out[(size_t)gm * 512 + (gn - 1024)] = v;
  }
}

// ---------------- Phase 2a: iou = iou_x[t] + h @ W_iouh^T; gates -------------
// Grid: 64 blocks x 256. Block = one 16-row tile x 128 cols (8 waves x 16 cols).
// Per k-slab: weight slab (128x32 bf16 = 8KB) DMA'd to LDS via async copies,
// h tile converted fp32->bf16 into LDS cooperatively (overlapped with the DMA).
__global__ void __launch_bounds__(256) gru_s1_k(
    const float*  __restrict__ h,       // [128][512] fp32 state
    const __bf16* __restrict__ Wbf,     // [1024][512] bf16 packed W_iouh
    const float*  __restrict__ b_iouh,  // [1024]
    const float*  __restrict__ iou_xt,  // [128][1024] slice for step t
    float* __restrict__ zbuf,           // [128][512]
    float* __restrict__ rhbuf)          // [128][512]  (r * h, fp32)
{
  __shared__ __align__(16) __bf16 Alds[16 * LDA_P];    // h tile (bf16)
  __shared__ __align__(16) __bf16 Wlds[128 * LDA_P];   // weight slab (bf16)

  const int tid   = threadIdx.x;
  const int wave  = tid >> 5;
  const int lane  = tid & 31;
  const int lhalf = lane >> 4;
  const int lrow  = lane & 15;

  const int rowBase      = (blockIdx.x >> 3) * 16;     // 0..112
  const int blockColBase = (blockIdx.x & 7) * 128;     // 0..896
  const unsigned wlds0   = (unsigned)(uintptr_t)(void*)&Wlds[0];

  v8f acc = {};
  for (int kb = 0; kb < MEM; kb += 32) {
    { // async DMA: 512 x 16B chunks of the weight slab, 2 chunks per thread
      const int c0 = tid, c1 = tid + 256;              // chunk: row=c>>2, quad=c&3
      const __bf16* g0 = Wbf + (size_t)(blockColBase + (c0 >> 2)) * MEM + kb + (c0 & 3) * 8;
      const __bf16* g1 = Wbf + (size_t)(blockColBase + (c1 >> 2)) * MEM + kb + (c1 & 3) * 8;
      async_copy_b128(wlds0 + (unsigned)((c0 >> 2) * (LDA_P * 2) + (c0 & 3) * 16), g0);
      async_copy_b128(wlds0 + (unsigned)((c1 >> 2) * (LDA_P * 2) + (c1 & 3) * 16), g1);
    }
    { // cooperative stage: h[rowBase..+16][kb..+32] -> bf16 LDS (overlaps DMA)
      const int e = tid * 2, m = e >> 5, k = e & 31;
      const float* hp = h + (size_t)(rowBase + m) * MEM + kb + k;
      Alds[m * LDA_P + k]     = (__bf16)hp[0];
      Alds[m * LDA_P + k + 1] = (__bf16)hp[1];
    }
    wait_async0();
    __syncthreads();
    v16bf a = frag_bf16(&Alds[lrow * LDA_P], 0, lhalf);               // ds_load_b128 x2
    v16bf b = frag_bf16(&Wlds[(wave * 16 + lrow) * LDA_P], 0, lhalf); // ds_load_b128 x2
    acc = WMMA_BF16(a, b, acc);
    __syncthreads();
  }

  const int   c    = blockColBase + wave * 16 + lrow;  // global output column
  const float bias = b_iouh[c];
#pragma unroll
  for (int r = 0; r < 8; ++r) {
    const int   row = rowBase + r + 8 * lhalf;
    const float v   = acc[r] + iou_xt[(size_t)row * 1024 + c] + bias;
    const float s   = sigmoid_f(v);
    if (c < 512) {
      zbuf[(size_t)row * MEM + c] = s;                 // z gate
    } else {
      const int cc = c - 512;                          // r gate -> r*h
      rhbuf[(size_t)row * MEM + cc] = s * h[(size_t)row * MEM + cc];
    }
  }
}

// ---------------- Phase 2b: h_tilde = tanh(f_x[t] + rh @ W_Uh^T); blend ------
// Grid: 32 blocks x 256. d_out[t] holds f_x[t] on entry, h[t] on exit (overlay).
__global__ void __launch_bounds__(256) gru_s2_k(
    float*        __restrict__ h,       // [128][512] fp32 state (in/out)
    const float*  __restrict__ rh,      // [128][512]
    const __bf16* __restrict__ Wbf,     // [512][512] bf16 packed W_Uh
    const float*  __restrict__ b_Uh,    // [512]
    const float*  __restrict__ zbuf,    // [128][512]
    float* __restrict__ fxt)            // d_out + t*B*M : f_x[t] in, h[t] out
{
  __shared__ __align__(16) __bf16 Alds[16 * LDA_P];
  __shared__ __align__(16) __bf16 Wlds[128 * LDA_P];

  const int tid   = threadIdx.x;
  const int wave  = tid >> 5;
  const int lane  = tid & 31;
  const int lhalf = lane >> 4;
  const int lrow  = lane & 15;

  const int rowBase      = (blockIdx.x >> 2) * 16;     // 0..112
  const int blockColBase = (blockIdx.x & 3) * 128;     // 0..384
  const unsigned wlds0   = (unsigned)(uintptr_t)(void*)&Wlds[0];

  v8f acc = {};
  for (int kb = 0; kb < MEM; kb += 32) {
    { // async DMA of W_Uh slab
      const int c0 = tid, c1 = tid + 256;
      const __bf16* g0 = Wbf + (size_t)(blockColBase + (c0 >> 2)) * MEM + kb + (c0 & 3) * 8;
      const __bf16* g1 = Wbf + (size_t)(blockColBase + (c1 >> 2)) * MEM + kb + (c1 & 3) * 8;
      async_copy_b128(wlds0 + (unsigned)((c0 >> 2) * (LDA_P * 2) + (c0 & 3) * 16), g0);
      async_copy_b128(wlds0 + (unsigned)((c1 >> 2) * (LDA_P * 2) + (c1 & 3) * 16), g1);
    }
    { // stage rh tile -> bf16 LDS
      const int e = tid * 2, m = e >> 5, k = e & 31;
      const float* ap = rh + (size_t)(rowBase + m) * MEM + kb + k;
      Alds[m * LDA_P + k]     = (__bf16)ap[0];
      Alds[m * LDA_P + k + 1] = (__bf16)ap[1];
    }
    wait_async0();
    __syncthreads();
    v16bf a = frag_bf16(&Alds[lrow * LDA_P], 0, lhalf);
    v16bf b = frag_bf16(&Wlds[(wave * 16 + lrow) * LDA_P], 0, lhalf);
    acc = WMMA_BF16(a, b, acc);
    __syncthreads();
  }

  const int   c    = blockColBase + wave * 16 + lrow;
  const float bias = b_Uh[c];
#pragma unroll
  for (int r = 0; r < 8; ++r) {
    const int    row = rowBase + r + 8 * lhalf;
    const size_t idx = (size_t)row * MEM + c;
    const float  u   = acc[r] + fxt[idx] + bias;       // read f_x[t] (overlay)
    const float  ht  = tanh_f(u);
    const float  z   = zbuf[idx];
    const float  hp  = h[idx];
    const float  hn  = (1.0f - z) * hp + z * ht;
    fxt[idx] = hn;                                     // write h[t] to d_out
    h[idx]   = hn;                                     // carry state
  }
}

// ---------------- host-side orchestration ------------------------------------
extern "C" void kernel_launch(void* const* d_in, const int* in_sizes, int n_in,
                              void* d_out, int out_size, void* d_ws, size_t ws_size,
                              hipStream_t stream) {
  (void)in_sizes; (void)n_in; (void)out_size; (void)ws_size;

  const float* X       = (const float*)d_in[0];  // [512,128,300]
  const float* W_ioux  = (const float*)d_in[1];  // [1024,300]
  const float* b_ioux  = (const float*)d_in[2];  // [1024]
  const float* W_iouh  = (const float*)d_in[3];  // [1024,512]
  const float* b_iouh  = (const float*)d_in[4];  // [1024]
  const float* W_fx    = (const float*)d_in[5];  // [512,300]
  const float* b_fx    = (const float*)d_in[6];  // [512]
  const float* W_Uh    = (const float*)d_in[7];  // [512,512]
  const float* b_Uh    = (const float*)d_in[8];  // [512]
  float*       out     = (float*)d_out;          // [512,128,512]

  // Workspace layout (f_x overlays d_out, so only iou_x is large):
  char*   ws        = (char*)d_ws;
  float*  iou_x     = (float*)ws;                                  // 256 MB
  size_t  off       = (size_t)NROWS * 1024 * sizeof(float);
  float*  hbuf      = (float*)(ws + off);        off += (size_t)BATCH * MEM * 4;
  float*  zbuf      = (float*)(ws + off);        off += (size_t)BATCH * MEM * 4;
  float*  rhbuf     = (float*)(ws + off);        off += (size_t)BATCH * MEM * 4;
  __bf16* Wiouh_bf  = (__bf16*)(ws + off);       off += (size_t)1024 * MEM * 2;
  __bf16* WUh_bf    = (__bf16*)(ws + off);

  // h0 = 0
  hipMemsetAsync(hbuf, 0, (size_t)BATCH * MEM * sizeof(float), stream);

  // Pack recurrent weights to bf16 once (L2-resident across the whole scan).
  pack_bf16_k<<<(1024 * MEM + 255) / 256, 256, 0, stream>>>(W_iouh, Wiouh_bf, 1024 * MEM);
  pack_bf16_k<<<(MEM * MEM + 255) / 256, 256, 0, stream>>>(W_Uh, WUh_bf, MEM * MEM);

  // Phase 1: iou_x -> ws, f_x -> d_out (overlay).
  // 4096 row-tiles x 96 col-tiles = 393216 wave-tiles / 8 waves per block.
  gemm_input_k<<<49152, 256, 0, stream>>>(X, W_ioux, b_ioux, W_fx, b_fx, iou_x, out);

  // Phase 2: sequential 512-step GRU scan; stream order carries the h dependency.
  for (int t = 0; t < TSTEPS; ++t) {
    gru_s1_k<<<64, 256, 0, stream>>>(hbuf, Wiouh_bf, b_iouh,
                                     iou_x + (size_t)t * BATCH * 1024, zbuf, rhbuf);
    gru_s2_k<<<32, 256, 0, stream>>>(hbuf, rhbuf, WUh_bf, b_Uh, zbuf,
                                     out + (size_t)t * BATCH * MEM);
  }
}
